// GRU_Full_3822520893655
// MI455X (gfx1250) — matrix operand order repaired and checked by reading, not compiled, not run on hardware
//
#include <hip/hip_runtime.h>
#include <hip/hip_bf16.h>

// Problem dimensions (fixed by the reference)
#define D_DIM 1024
#define B_SZ  128
#define VOUT  8000
#define SLEN  64
#define TLEN  64

typedef __attribute__((ext_vector_type(16))) __bf16       bf16x16;
typedef __attribute__((ext_vector_type(8)))  float        f32x8;
typedef __attribute__((ext_vector_type(4)))  float        f32x4;
typedef __attribute__((ext_vector_type(4)))  unsigned int u32x4;

// Pack two f32 into one dword of two bf16 (truncation) with a single v_perm_b32:
// result = {hi[31:16], lo[31:16]}; sel bytes: {7,6,3,2}.
__device__ __forceinline__ unsigned int pk_bf16(float lo, float hi) {
  return __builtin_amdgcn_perm(__float_as_uint(hi), __float_as_uint(lo), 0x07060302u);
}

// ---- B-tile loaders (one lane's 16 contiguous K values of a 32x16 B tile) ----
__device__ __forceinline__ bf16x16 load_b_tile(const unsigned short* __restrict__ p) {
  union { u32x4 q[2]; bf16x16 v; } u;
  u.q[0] = *(const u32x4*)(p);      // K elems 0..7  (16 B)
  u.q[1] = *(const u32x4*)(p + 8);  // K elems 8..15 (16 B)
  return u.v;
}
__device__ __forceinline__ bf16x16 load_b_tile(const float* __restrict__ p) {
  f32x4 w0 = *(const f32x4*)(p);
  f32x4 w1 = *(const f32x4*)(p + 4);
  f32x4 w2 = *(const f32x4*)(p + 8);
  f32x4 w3 = *(const f32x4*)(p + 12);
  union { unsigned int d[8]; bf16x16 v; } u;
  u.d[0] = pk_bf16(w0[0], w0[1]); u.d[1] = pk_bf16(w0[2], w0[3]);
  u.d[2] = pk_bf16(w1[0], w1[1]); u.d[3] = pk_bf16(w1[2], w1[3]);
  u.d[4] = pk_bf16(w2[0], w2[1]); u.d[5] = pk_bf16(w2[2], w2[3]);
  u.d[6] = pk_bf16(w3[0], w3[1]); u.d[7] = pk_bf16(w3[2], w3[3]);
  return u.v;
}

// One wave computes a 16(M) x 64(N) tile of out = A @ W^T + bias.
// A: [128,1024] bf16 (row-major), W: [N,1024] (row-major, bf16 or f32), K=1024.
// Wave = M-tile (threadIdx.x>>5), nq = N/64 column quad.
template <typename WT>
__device__ __forceinline__ void gemm_wave(
    const unsigned short* __restrict__ A,
    const WT* __restrict__ W,
    const float* __restrict__ bias,
    float* __restrict__ out,
    int N, int nq)
{
  const int K    = D_DIM;
  const int lane = threadIdx.x & 31;
  const int mt   = threadIdx.x >> 5;
  const int half = lane >> 4;
  const int l16  = lane & 15;

  // A layout: lane holds row l16; 16 contiguous K values
  // starting at K = k0 + 16*half (matches B).
  const unsigned short* arow = A + (size_t)(mt * 16 + l16) * K + half * 16;

  // B layout (16-bit B 32x16): lane holds column l16; 16 contiguous K values
  // starting at K = k0 + 16*half.
  const WT* w0p = W + (size_t)(nq * 64 +  0 + l16) * K + half * 16;
  const WT* w1p = W + (size_t)(nq * 64 + 16 + l16) * K + half * 16;
  const WT* w2p = W + (size_t)(nq * 64 + 32 + l16) * K + half * 16;
  const WT* w3p = W + (size_t)(nq * 64 + 48 + l16) * K + half * 16;

  f32x8 acc0 = {}, acc1 = {}, acc2 = {}, acc3 = {};

  for (int k0 = 0; k0 < K; k0 += 32) {
    union { u32x4 q[2]; bf16x16 v; } au;
    au.q[0] = *(const u32x4*)(arow + k0);
    au.q[1] = *(const u32x4*)(arow + k0 + 8);

    bf16x16 b0 = load_b_tile(w0p + k0);
    acc0 = __builtin_amdgcn_wmma_f32_16x16x32_bf16(false, au.v, false, b0,
                                                   (short)0, acc0, false, false);
    bf16x16 b1 = load_b_tile(w1p + k0);
    acc1 = __builtin_amdgcn_wmma_f32_16x16x32_bf16(false, au.v, false, b1,
                                                   (short)0, acc1, false, false);
    bf16x16 b2 = load_b_tile(w2p + k0);
    acc2 = __builtin_amdgcn_wmma_f32_16x16x32_bf16(false, au.v, false, b2,
                                                   (short)0, acc2, false, false);
    bf16x16 b3 = load_b_tile(w3p + k0);
    acc3 = __builtin_amdgcn_wmma_f32_16x16x32_bf16(false, au.v, false, b3,
                                                   (short)0, acc3, false, false);
  }

  // C/D layout: c[v] = out[mt*16 + 8*half + v, nq*64 + sub*16 + l16]
  const int nbase = nq * 64 + l16;
  const float bv0 = bias[nbase];
  const float bv1 = bias[nbase + 16];
  const float bv2 = bias[nbase + 32];
  const float bv3 = bias[nbase + 48];
  float* orow = out + (size_t)(mt * 16 + half * 8) * N + nbase;
#pragma unroll
  for (int v = 0; v < 8; ++v) {
    orow[(size_t)v * N +  0] = acc0[v] + bv0;
    orow[(size_t)v * N + 16] = acc1[v] + bv1;
    orow[(size_t)v * N + 32] = acc2[v] + bv2;
    orow[(size_t)v * N + 48] = acc3[v] + bv3;
  }
}

// Generic GEMM launch wrapper (N/64 blocks x 8 waves).
template <typename WT>
__global__ __launch_bounds__(256) void gemm_x_wT_bias(
    const unsigned short* __restrict__ A, const WT* __restrict__ W,
    const float* __restrict__ bias, float* __restrict__ out, int N)
{
  gemm_wave(A, W, bias, out, N, blockIdx.x);
}

// Fused GRU gate GEMMs: gi = x@Wih^T+bih and gh = h@Whh^T+bhh run concurrently.
// Grid: 96 blocks. Blocks 0..47 -> gi, 48..95 -> gh (scalar branch; EXEC stays full).
template <typename WT>
__global__ __launch_bounds__(256) void gru_gates(
    const unsigned short* __restrict__ xb, const WT* __restrict__ Wih,
    const float* __restrict__ bih, float* __restrict__ gi,
    const unsigned short* __restrict__ hb, const WT* __restrict__ Whh,
    const float* __restrict__ bhh, float* __restrict__ gh)
{
  const int NQ = (3 * D_DIM) / 64;  // 48
  int nq = blockIdx.x;
  if (nq < NQ) {
    gemm_wave(xb, Wih, bih, gi, 3 * D_DIM, nq);
  } else {
    gemm_wave(hb, Whh, bhh, gh, 3 * D_DIM, nq - NQ);
  }
}

// One-time f32 -> bf16 weight conversion; 8 elems/thread, n % 2048 == 0.
__global__ __launch_bounds__(256) void convert_f32_bf16(
    const float* __restrict__ in, unsigned short* __restrict__ outw)
{
  size_t i = ((size_t)blockIdx.x * 256 + threadIdx.x) * 8;
  f32x4 a = *(const f32x4*)(in + i);
  f32x4 b = *(const f32x4*)(in + i + 4);
  u32x4 o;
  o[0] = pk_bf16(a[0], a[1]); o[1] = pk_bf16(a[2], a[3]);
  o[2] = pk_bf16(b[0], b[1]); o[3] = pk_bf16(b[2], b[3]);
  *(u32x4*)(outw + i) = o;
}

// x_bf16[b, :] = bf16(emb[tokens[b], :])
__global__ __launch_bounds__(256) void embed_bf16(
    const int* __restrict__ tokens, const float* __restrict__ emb,
    unsigned short* __restrict__ xb)
{
  int i = blockIdx.x * 256 + threadIdx.x;   // 0 .. 131071
  int b = i >> 10;
  int j = i & (D_DIM - 1);
  int t = tokens[b];
  float v = emb[(size_t)t * D_DIM + j];
  xb[i] = (unsigned short)(__float_as_uint(v) >> 16);
}

// GRU gate fusion (PyTorch order r,z,n): h_new = (1-z)*n + z*h
__global__ __launch_bounds__(256) void gru_fuse(
    const float* __restrict__ gi, const float* __restrict__ gh,
    float* __restrict__ h, unsigned short* __restrict__ hb)
{
  int i = blockIdx.x * 256 + threadIdx.x;   // 0 .. 131071
  int b = i >> 10;
  int j = i & (D_DIM - 1);
  const float* gib = gi + (size_t)b * 3 * D_DIM;
  const float* ghb = gh + (size_t)b * 3 * D_DIM;
  float ir = gib[j], iz = gib[D_DIM + j], inn = gib[2 * D_DIM + j];
  float hr = ghb[j], hz = ghb[D_DIM + j], hn  = ghb[2 * D_DIM + j];
  float r = 1.0f / (1.0f + __expf(-(ir + hr)));
  float z = 1.0f / (1.0f + __expf(-(iz + hz)));
  float n = tanhf(inn + r * hn);
  float hold = h[i];
  float hnew = (1.0f - z) * n + z * hold;
  h[i]  = hnew;
  hb[i] = (unsigned short)(__float_as_uint(hnew) >> 16);
}

__global__ __launch_bounds__(256) void init_h(
    float* __restrict__ h, unsigned short* __restrict__ hb)
{
  int i = blockIdx.x * 256 + threadIdx.x;
  h[i]  = 0.0f;
  hb[i] = 0;
}

// Row-wise argmax over 8000 logits; first-max tie-break (matches jnp.argmax).
__global__ __launch_bounds__(256) void argmax_row(
    const float* __restrict__ logits, int* __restrict__ tok)
{
  const float* row = logits + (size_t)blockIdx.x * VOUT;
  float best = -__builtin_inff();
  int   bi   = 0x7fffffff;
  for (int j = threadIdx.x; j < VOUT; j += 256) {
    float v = row[j];
    if (v > best || (v == best && j < bi)) { best = v; bi = j; }
  }
  __shared__ float sv[256];
  __shared__ int   si[256];
  sv[threadIdx.x] = best;
  si[threadIdx.x] = bi;
  __syncthreads();
  for (int s = 128; s > 0; s >>= 1) {
    if (threadIdx.x < s) {
      float v2 = sv[threadIdx.x + s];
      int   i2 = si[threadIdx.x + s];
      if (v2 > sv[threadIdx.x] ||
          (v2 == sv[threadIdx.x] && i2 < si[threadIdx.x])) {
        sv[threadIdx.x] = v2;
        si[threadIdx.x] = i2;
      }
    }
    __syncthreads();
  }
  if (threadIdx.x == 0) tok[blockIdx.x] = si[0];
}

// ---- host-side model driver, templated on weight storage type ----
template <typename WT>
static void run_model(const int* src, const int* tgt,
                      const float* enc_emb, const WT* enc_Wih, const WT* enc_Whh,
                      const float* enc_bih, const float* enc_bhh,
                      const float* dec_emb, const WT* dec_Wih, const WT* dec_Whh,
                      const float* dec_bih, const float* dec_bhh,
                      const WT* W_out, const float* b_out,
                      float* out, float* gi, float* gh, float* h,
                      unsigned short* hb, unsigned short* xb, int* tok,
                      hipStream_t stream)
{
  init_h<<<512, 256, 0, stream>>>(h, hb);

  // ---- encoder ----
  for (int t = 0; t < SLEN; ++t) {
    embed_bf16<<<512, 256, 0, stream>>>(src + t * B_SZ, enc_emb, xb);
    gru_gates<WT><<<96, 256, 0, stream>>>(xb, enc_Wih, enc_bih, gi,
                                          hb, enc_Whh, enc_bhh, gh);
    gru_fuse<<<512, 256, 0, stream>>>(gi, gh, h, hb);
  }

  // ---- decoder (greedy) ----
  for (int t = 0; t < TLEN; ++t) {
    const int* toks = (t == 0) ? tgt : tok;   // tgt[0] row = first 128 ints
    embed_bf16<<<512, 256, 0, stream>>>(toks, dec_emb, xb);
    gru_gates<WT><<<96, 256, 0, stream>>>(xb, dec_Wih, dec_bih, gi,
                                          hb, dec_Whh, dec_bhh, gh);
    gru_fuse<<<512, 256, 0, stream>>>(gi, gh, h, hb);
    float* lg = out + (size_t)t * B_SZ * VOUT;
    gemm_x_wT_bias<WT><<<VOUT / 64, 256, 0, stream>>>(hb, W_out, b_out, lg, VOUT);
    argmax_row<<<B_SZ, 256, 0, stream>>>(lg, tok);
  }
}

extern "C" void kernel_launch(void* const* d_in, const int* in_sizes, int n_in,
                              void* d_out, int out_size, void* d_ws, size_t ws_size,
                              hipStream_t stream) {
  const int*   src     = (const int*)  d_in[0];
  const int*   tgt     = (const int*)  d_in[1];
  const float* enc_emb = (const float*)d_in[2];
  const float* enc_Wih = (const float*)d_in[3];
  const float* enc_Whh = (const float*)d_in[4];
  const float* enc_bih = (const float*)d_in[5];
  const float* enc_bhh = (const float*)d_in[6];
  const float* dec_emb = (const float*)d_in[7];
  const float* dec_Wih = (const float*)d_in[8];
  const float* dec_Whh = (const float*)d_in[9];
  const float* dec_bih = (const float*)d_in[10];
  const float* dec_bhh = (const float*)d_in[11];
  const float* W_out   = (const float*)d_in[12];
  const float* b_out   = (const float*)d_in[13];
  float* out = (float*)d_out;

  const size_t GRU_W  = (size_t)3 * D_DIM * D_DIM;  // 3,145,728 elems
  const size_t OUT_W  = (size_t)VOUT * D_DIM;       // 8,192,000 elems
  const size_t W_B16  = (4 * GRU_W + OUT_W) * sizeof(unsigned short);
  const size_t ACT_B  = (size_t)B_SZ * 3 * D_DIM * 4 * 2   // gi, gh
                      + (size_t)B_SZ * D_DIM * 4           // h (f32)
                      + (size_t)B_SZ * D_DIM * 2 * 2       // hb, xb (bf16)
                      + 512;                               // tok
  const bool fast = ws_size >= W_B16 + ACT_B;

  char* ws = (char*)d_ws;
  unsigned short* wb = (unsigned short*)ws;
  unsigned short* encWihB = wb;
  unsigned short* encWhhB = encWihB + GRU_W;
  unsigned short* decWihB = encWhhB + GRU_W;
  unsigned short* decWhhB = decWihB + GRU_W;
  unsigned short* WoutB   = decWhhB + GRU_W;
  char* act = fast ? (char*)(WoutB + OUT_W) : ws;

  float* gi = (float*)act;                                  // [128, 3072]
  float* gh = gi + (size_t)B_SZ * 3 * D_DIM;                // [128, 3072]
  float* h  = gh + (size_t)B_SZ * 3 * D_DIM;                // [128, 1024] f32
  unsigned short* hb = (unsigned short*)(h + (size_t)B_SZ * D_DIM);
  unsigned short* xb = hb + (size_t)B_SZ * D_DIM;
  int* tok = (int*)(xb + (size_t)B_SZ * D_DIM);

  if (fast) {
    // One-time weight conversion (amortized over 128 steps; bf16 weights
    // total ~41.6 MB and stay resident in the 192 MB L2).
    convert_f32_bf16<<<GRU_W / 2048, 256, 0, stream>>>(enc_Wih, encWihB);
    convert_f32_bf16<<<GRU_W / 2048, 256, 0, stream>>>(enc_Whh, encWhhB);
    convert_f32_bf16<<<GRU_W / 2048, 256, 0, stream>>>(dec_Wih, decWihB);
    convert_f32_bf16<<<GRU_W / 2048, 256, 0, stream>>>(dec_Whh, decWhhB);
    convert_f32_bf16<<<OUT_W / 2048, 256, 0, stream>>>(W_out, WoutB);
    run_model<unsigned short>(src, tgt, enc_emb, encWihB, encWhhB, enc_bih, enc_bhh,
                              dec_emb, decWihB, decWhhB, dec_bih, dec_bhh,
                              WoutB, b_out, out, gi, gh, h, hb, xb, tok, stream);
  } else {
    run_model<float>(src, tgt, enc_emb, enc_Wih, enc_Whh, enc_bih, enc_bhh,
                     dec_emb, dec_Wih, dec_Whh, dec_bih, dec_bhh,
                     W_out, b_out, out, gi, gh, h, hb, xb, tok, stream);
  }
}
